// PatchCoreModel_56238301774105
// MI455X (gfx1250) — compile-verified, matrix operand order checked by reading, not run on hardware
//
#include <hip/hip_runtime.h>
#include <hip/hip_bf16.h>

typedef _Float16 v16h __attribute__((ext_vector_type(16)));
typedef float    v8f  __attribute__((ext_vector_type(8)));

union Frag16 { v16h v; uint4 u[2]; };

#define EPS_BN   1e-5f
#define EPS_NORM 1e-12f

// ---------------------------------------------------------------------------
// Memory-bank row L2-normalize -> f16.  8 rows per 256-thread block,
// one wave (32 lanes) per row, 2 floats per lane.
// ---------------------------------------------------------------------------
__global__ __launch_bounds__(256) void mb_norm_kernel(const float* __restrict__ mb,
                                                      _Float16* __restrict__ mbh,
                                                      int nrows) {
    int row  = blockIdx.x * 8 + (threadIdx.x >> 5);
    int lane = threadIdx.x & 31;
    if (row >= nrows) return;
    const float2 v = *(const float2*)(mb + (size_t)row * 64 + lane * 2);
    float ss = v.x * v.x + v.y * v.y;
#pragma unroll
    for (int m = 16; m >= 1; m >>= 1) ss += __shfl_xor(ss, m, 32);
    float scale = 1.0f / fmaxf(sqrtf(ss), EPS_NORM);
    _Float16* dst = mbh + (size_t)row * 64 + lane * 2;
    dst[0] = (_Float16)(v.x * scale);
    dst[1] = (_Float16)(v.y * scale);
}

// ---------------------------------------------------------------------------
// conv1: [256,1,64,64] -> [256,16,32,32], 3x3 s2 p1, fused BN+ReLU
// ---------------------------------------------------------------------------
__global__ __launch_bounds__(256) void conv1_kernel(const float* __restrict__ x,
        const float* __restrict__ w, const float* __restrict__ b,
        const float* __restrict__ gamma, const float* __restrict__ beta,
        const float* __restrict__ mean, const float* __restrict__ var,
        float* __restrict__ out) {
    int idx = blockIdx.x * 256 + threadIdx.x;           // 256*16*32*32
    int ox = idx & 31, oy = (idx >> 5) & 31, co = (idx >> 10) & 15, bb = idx >> 14;
    const float* xb = x + (size_t)bb * 64 * 64;
    const float* wc = w + co * 9;
    float acc = b[co];
    int iy0 = oy * 2 - 1, ix0 = ox * 2 - 1;
#pragma unroll
    for (int ky = 0; ky < 3; ++ky) {
        int iy = iy0 + ky;
        if (iy < 0 || iy >= 64) continue;
#pragma unroll
        for (int kx = 0; kx < 3; ++kx) {
            int ix = ix0 + kx;
            if (ix < 0 || ix >= 64) continue;
            acc += xb[iy * 64 + ix] * wc[ky * 3 + kx];
        }
    }
    float s = gamma[co] * rsqrtf(var[co] + EPS_BN);
    float y = (acc - mean[co]) * s + beta[co];
    out[idx] = fmaxf(y, 0.0f);
}

// ---------------------------------------------------------------------------
// conv2: [256,16,32,32] -> [256,32,16,16], fused BN+ReLU
// ---------------------------------------------------------------------------
__global__ __launch_bounds__(256) void conv2_kernel(const float* __restrict__ h1,
        const float* __restrict__ w, const float* __restrict__ b,
        const float* __restrict__ gamma, const float* __restrict__ beta,
        const float* __restrict__ mean, const float* __restrict__ var,
        float* __restrict__ out) {
    int idx = blockIdx.x * 256 + threadIdx.x;           // 256*32*16*16
    int ox = idx & 15, oy = (idx >> 4) & 15, co = (idx >> 8) & 31, bb = idx >> 13;
    const float* xb = h1 + (size_t)bb * 16 * 1024;
    float acc = b[co];
    int iy0 = oy * 2 - 1, ix0 = ox * 2 - 1;
    for (int ci = 0; ci < 16; ++ci) {
        const float* xc = xb + ci * 1024;
        const float* wc = w + (co * 16 + ci) * 9;
#pragma unroll
        for (int ky = 0; ky < 3; ++ky) {
            int iy = iy0 + ky;
            if (iy < 0 || iy >= 32) continue;
#pragma unroll
            for (int kx = 0; kx < 3; ++kx) {
                int ix = ix0 + kx;
                if (ix < 0 || ix >= 32) continue;
                acc += xc[iy * 32 + ix] * wc[ky * 3 + kx];
            }
        }
    }
    float s = gamma[co] * rsqrtf(var[co] + EPS_BN);
    float y = (acc - mean[co]) * s + beta[co];
    out[idx] = fmaxf(y, 0.0f);
}

// ---------------------------------------------------------------------------
// conv3: [256,32,16,16] -> queries q[(b*64 + y*8 + x)][c], fused BN+ReLU
// ---------------------------------------------------------------------------
__global__ __launch_bounds__(256) void conv3_kernel(const float* __restrict__ h2,
        const float* __restrict__ w, const float* __restrict__ b,
        const float* __restrict__ gamma, const float* __restrict__ beta,
        const float* __restrict__ mean, const float* __restrict__ var,
        float* __restrict__ q) {
    int idx = blockIdx.x * 256 + threadIdx.x;           // 256*64*8*8
    int ox = idx & 7, oy = (idx >> 3) & 7, co = (idx >> 6) & 63, bb = idx >> 12;
    const float* xb = h2 + (size_t)bb * 32 * 256;
    float acc = b[co];
    int iy0 = oy * 2 - 1, ix0 = ox * 2 - 1;
    for (int ci = 0; ci < 32; ++ci) {
        const float* xc = xb + ci * 256;
        const float* wc = w + (co * 32 + ci) * 9;
#pragma unroll
        for (int ky = 0; ky < 3; ++ky) {
            int iy = iy0 + ky;
            if (iy < 0 || iy >= 16) continue;
#pragma unroll
            for (int kx = 0; kx < 3; ++kx) {
                int ix = ix0 + kx;
                if (ix < 0 || ix >= 16) continue;
                acc += xc[iy * 16 + ix] * wc[ky * 3 + kx];
            }
        }
    }
    float s = gamma[co] * rsqrtf(var[co] + EPS_BN);
    float y = (acc - mean[co]) * s + beta[co];
    q[(size_t)(bb * 64 + oy * 8 + ox) * 64 + co] = fmaxf(y, 0.0f);
}

// ---------------------------------------------------------------------------
// Query row L2-normalize -> f16 + q_sq (sum of squares of normalized row)
// ---------------------------------------------------------------------------
__global__ __launch_bounds__(256) void q_norm_kernel(const float* __restrict__ q,
                                                     _Float16* __restrict__ qh,
                                                     float* __restrict__ qsq,
                                                     int nrows) {
    int row  = blockIdx.x * 8 + (threadIdx.x >> 5);
    int lane = threadIdx.x & 31;
    if (row >= nrows) return;
    const float2 v = *(const float2*)(q + (size_t)row * 64 + lane * 2);
    float ss = v.x * v.x + v.y * v.y;
#pragma unroll
    for (int m = 16; m >= 1; m >>= 1) ss += __shfl_xor(ss, m, 32);
    float scale = 1.0f / fmaxf(sqrtf(ss), EPS_NORM);
    _Float16* dst = qh + (size_t)row * 64 + lane * 2;
    dst[0] = (_Float16)(v.x * scale);
    dst[1] = (_Float16)(v.y * scale);
    if (lane == 0) qsq[row] = ss * scale * scale;
}

// ---------------------------------------------------------------------------
// Main WMMA kernel.  8 waves/block; each wave owns TWO 16-query A tiles
// (reuses every LDS B fragment twice -> 16 WMMA per 16 ds_load_b128).
// Memory bank split across gridDim.y; LDS chunk double-buffered so global
// loads for chunk i+1 overlap WMMA compute on chunk i (1 barrier/iter).
// Writes per-split max cosine sim.
// ---------------------------------------------------------------------------
#define MCHUNK 64   // memory rows per LDS chunk (4 N-tiles of 16)
#define NSPLIT 4

__global__ __launch_bounds__(256) void maxsim_kernel(const _Float16* __restrict__ qh,
                                                     const _Float16* __restrict__ mbh,
                                                     float* __restrict__ psim,
                                                     int nmem, int nq) {
    __shared__ __align__(16) _Float16 smem[2][MCHUNK * 64];   // 2 x 8 KB

    const int wave = threadIdx.x >> 5;
    const int lane = threadIdx.x & 31;
    const int mrow = lane & 15;          // M (A) or N (B) index within tile
    const int g    = lane >> 4;          // lane group selects K sub-range

    const int qbaseA = (blockIdx.x * 8 + wave) * 32;   // two adjacent 16-query tiles
    const int qbaseB = qbaseA + 16;

    // A fragments (16x32 f16 layout): lane l holds row l%16,
    // K-halves {seg*16 + g*8 + 0..7}; frags 0/1 cover K=0..31 / 32..63.
    Frag16 a0A, a1A, a0B, a1B;
    {
        const _Float16* qrow = qh + (size_t)(qbaseA + mrow) * 64;
        a0A.u[0] = *(const uint4*)(qrow +      g * 8);
        a0A.u[1] = *(const uint4*)(qrow + 16 + g * 8);
        a1A.u[0] = *(const uint4*)(qrow + 32 + g * 8);
        a1A.u[1] = *(const uint4*)(qrow + 48 + g * 8);
        qrow += 16 * 64;
        a0B.u[0] = *(const uint4*)(qrow +      g * 8);
        a0B.u[1] = *(const uint4*)(qrow + 16 + g * 8);
        a1B.u[0] = *(const uint4*)(qrow + 32 + g * 8);
        a1B.u[1] = *(const uint4*)(qrow + 48 + g * 8);
    }

    v8f rmA, rmB;
#pragma unroll
    for (int i = 0; i < 8; ++i) { rmA[i] = -3.0e38f; rmB[i] = -3.0e38f; }

    const int mper  = nmem / gridDim.y;        // bank rows per split
    const int mbase = blockIdx.y * mper;
    const int niter = mper / MCHUNK;

    // prologue: stage chunk 0 into buffer 0
    {
        const uint4* src = (const uint4*)(mbh + (size_t)mbase * 64);
        uint4* dst = (uint4*)smem[0];
        dst[threadIdx.x]       = src[threadIdx.x];
        dst[threadIdx.x + 256] = src[threadIdx.x + 256];
    }
    __syncthreads();

    for (int it = 0; it < niter; ++it) {
        uint4 r0 = {}, r1 = {};
        const bool has_next = (it + 1 < niter);
        if (has_next) {
            const uint4* src = (const uint4*)(mbh + (size_t)(mbase + (it + 1) * MCHUNK) * 64);
            r0 = src[threadIdx.x];
            r1 = src[threadIdx.x + 256];
            if (it + 2 < niter)
                __builtin_prefetch(mbh + (size_t)(mbase + (it + 2) * MCHUNK) * 64 + threadIdx.x * 16, 0, 1);
        }

        const _Float16* buf = smem[it & 1];
#pragma unroll
        for (int t = 0; t < MCHUNK / 16; ++t) {
            const _Float16* brow = buf + (t * 16 + mrow) * 64;
            Frag16 b0, b1;
            b0.u[0] = *(const uint4*)(brow +      g * 8);
            b0.u[1] = *(const uint4*)(brow + 16 + g * 8);
            b1.u[0] = *(const uint4*)(brow + 32 + g * 8);
            b1.u[1] = *(const uint4*)(brow + 48 + g * 8);

            v8f cA = {};
            cA = __builtin_amdgcn_wmma_f32_16x16x32_f16(false, a0A.v, false, b0.v,
                                                        (short)0, cA, false, false);
            cA = __builtin_amdgcn_wmma_f32_16x16x32_f16(false, a1A.v, false, b1.v,
                                                        (short)0, cA, false, false);
            v8f cB = {};
            cB = __builtin_amdgcn_wmma_f32_16x16x32_f16(false, a0B.v, false, b0.v,
                                                        (short)0, cB, false, false);
            cB = __builtin_amdgcn_wmma_f32_16x16x32_f16(false, a1B.v, false, b1.v,
                                                        (short)0, cB, false, false);
#pragma unroll
            for (int i = 0; i < 8; ++i) {
                rmA[i] = fmaxf(rmA[i], cA[i]);
                rmB[i] = fmaxf(rmB[i], cB[i]);
            }
        }

        if (has_next) {   // write NEXT chunk into the other buffer (nobody reads it now)
            uint4* dst = (uint4*)smem[(it + 1) & 1];
            dst[threadIdx.x]       = r0;
            dst[threadIdx.x + 256] = r1;
        }
        __syncthreads();
    }

    // reduce max over N (16 lanes within each half-wave group)
#pragma unroll
    for (int m = 8; m >= 1; m >>= 1)
#pragma unroll
        for (int i = 0; i < 8; ++i) {
            rmA[i] = fmaxf(rmA[i], __shfl_xor(rmA[i], m, 32));
            rmB[i] = fmaxf(rmB[i], __shfl_xor(rmB[i], m, 32));
        }

    if (mrow == 0) {
        float* row = psim + (size_t)blockIdx.y * nq;
#pragma unroll
        for (int v = 0; v < 8; ++v) {          // D layout: (M = v + 8*(l/16), N = l%16)
            row[qbaseA + v + 8 * g] = rmA[v];
            row[qbaseB + v + 8 * g] = rmB[v];
        }
    }
}

// ---------------------------------------------------------------------------
// Combine splits: max sim over splits -> per-patch distance
// ---------------------------------------------------------------------------
__global__ __launch_bounds__(256) void combine_kernel(const float* __restrict__ psim,
                                                      const float* __restrict__ qsq,
                                                      float* __restrict__ pdist,
                                                      int nsplit, int nq) {
    int q = blockIdx.x * 256 + threadIdx.x;
    if (q >= nq) return;
    float s = -3.0e38f;
    for (int i = 0; i < nsplit; ++i) s = fmaxf(s, psim[(size_t)i * nq + q]);
    float d2 = qsq[q] + 1.0f - 2.0f * s;
    pdist[q] = sqrtf(fmaxf(d2, 0.0f));
}

// ---------------------------------------------------------------------------
// Per-image max over 64 patches -> out[256]
// ---------------------------------------------------------------------------
__global__ __launch_bounds__(32) void image_max_kernel(const float* __restrict__ pdist,
                                                       float* __restrict__ out) {
    int b = blockIdx.x, lane = threadIdx.x;
    float m = fmaxf(pdist[b * 64 + lane], pdist[b * 64 + 32 + lane]);
#pragma unroll
    for (int o = 16; o >= 1; o >>= 1) m = fmaxf(m, __shfl_xor(m, o, 32));
    if (lane == 0) out[b] = m;
}

// ---------------------------------------------------------------------------
extern "C" void kernel_launch(void* const* d_in, const int* in_sizes, int n_in,
                              void* d_out, int out_size, void* d_ws, size_t ws_size,
                              hipStream_t stream) {
    const float* x  = (const float*)d_in[0];
    const float* mb = (const float*)d_in[1];
    const float* w0 = (const float*)d_in[2];
    const float* b0 = (const float*)d_in[3];
    const float* g0 = (const float*)d_in[4];
    const float* t0 = (const float*)d_in[5];
    const float* m0 = (const float*)d_in[6];
    const float* v0 = (const float*)d_in[7];
    const float* w1 = (const float*)d_in[8];
    const float* b1 = (const float*)d_in[9];
    const float* g1 = (const float*)d_in[10];
    const float* t1 = (const float*)d_in[11];
    const float* m1 = (const float*)d_in[12];
    const float* v1 = (const float*)d_in[13];
    const float* w2 = (const float*)d_in[14];
    const float* b2 = (const float*)d_in[15];
    const float* g2 = (const float*)d_in[16];
    const float* t2 = (const float*)d_in[17];
    const float* m2 = (const float*)d_in[18];
    const float* v2 = (const float*)d_in[19];

    const int nmem = in_sizes[1] / 64;           // 262144
    const int NQ = 16384;                        // 256 images * 64 patches

    char* ws = (char*)d_ws;
    size_t off = 0;
    _Float16* mbh  = (_Float16*)(ws + off); off += (size_t)nmem * 64 * 2;   // 32 MB
    float*    h1   = (float*)(ws + off);    off += (size_t)256 * 16 * 32 * 32 * 4;
    float*    h2   = (float*)(ws + off);    off += (size_t)256 * 32 * 16 * 16 * 4;
    float*    qf   = (float*)(ws + off);    off += (size_t)NQ * 64 * 4;
    _Float16* qh   = (_Float16*)(ws + off); off += (size_t)NQ * 64 * 2;
    float*    qsq  = (float*)(ws + off);    off += (size_t)NQ * 4;
    float*    psim = (float*)(ws + off);    off += (size_t)NSPLIT * NQ * 4;
    float*    pd   = (float*)(ws + off);    off += (size_t)NQ * 4;

    mb_norm_kernel<<<nmem / 8, 256, 0, stream>>>(mb, mbh, nmem);
    conv1_kernel<<<(256 * 16 * 32 * 32) / 256, 256, 0, stream>>>(x,  w0, b0, g0, t0, m0, v0, h1);
    conv2_kernel<<<(256 * 32 * 16 * 16) / 256, 256, 0, stream>>>(h1, w1, b1, g1, t1, m1, v1, h2);
    conv3_kernel<<<(256 * 64 *  8 *  8) / 256, 256, 0, stream>>>(h2, w2, b2, g2, t2, m2, v2, qf);
    q_norm_kernel<<<NQ / 8, 256, 0, stream>>>(qf, qh, qsq, NQ);
    maxsim_kernel<<<dim3(NQ / 256, NSPLIT), 256, 0, stream>>>(qh, mbh, psim, nmem, NQ);
    combine_kernel<<<(NQ + 255) / 256, 256, 0, stream>>>(psim, qsq, pd, NSPLIT, NQ);
    image_max_kernel<<<256, 32, 0, stream>>>(pd, (float*)d_out);
}
